// CombinedLoss_21036749815896
// MI455X (gfx1250) — matrix operand order repaired and checked by reading, not compiled, not run on hardware
//
#include <hip/hip_runtime.h>
#include <hip/hip_bf16.h>

typedef __attribute__((ext_vector_type(8)))  float  v8f;
typedef __attribute__((ext_vector_type(16))) __bf16 v16bf;
typedef int v4i_ __attribute__((vector_size(16)));
typedef __attribute__((address_space(1))) v4i_* gv4p;
typedef __attribute__((address_space(3))) v4i_* lv4p;

// ---------- bf16 helpers ----------
static __device__ inline unsigned short f2bf(float f) {
  unsigned u = __builtin_bit_cast(unsigned, f);
  unsigned r = u + 0x7FFFu + ((u >> 16) & 1u);
  return (unsigned short)(r >> 16);
}
static __device__ inline float bf2f(unsigned short h) {
  unsigned u = ((unsigned)h) << 16;
  return __builtin_bit_cast(float, u);
}

// ---------- block reduction (blockDim.x == 256) ----------
static __device__ inline float block_reduce256(float v) {
  __shared__ float s[256];
  __syncthreads();
  s[threadIdx.x] = v;
  __syncthreads();
  for (int off = 128; off > 0; off >>= 1) {
    if ((int)threadIdx.x < off) s[threadIdx.x] += s[threadIdx.x + off];
    __syncthreads();
  }
  float r = s[0];
  __syncthreads();
  return r;
}

// ---------- zero scalar/hist region ----------
__global__ void zero_k(float* S) {
  for (int i = threadIdx.x; i < 1024; i += 256) S[i] = 0.0f;
}

// ---------- weight repack: f32 OIHW -> bf16 [tap][cout][cin] ----------
__global__ void pack_w_k(const float* __restrict__ W, unsigned short* __restrict__ out,
                         int cin, int cout, long long total) {
  long long tid = blockIdx.x * 256LL + threadIdx.x;
  if (tid >= total) return;
  long long oc = (long long)cout * cin;
  int tap = (int)(tid / oc);
  long long r = tid - (long long)tap * oc;
  int o = (int)(r / cin);
  int i = (int)(r - (long long)o * cin);
  out[tid] = f2bf(W[((long long)o * cin + i) * 9 + tap]);
}

// ---------- first conv: cin=3, cout=64, f32 NCHW -> bf16 NHWC, bias+relu ----------
__global__ void conv_first_k(const float* __restrict__ in, const float* __restrict__ Wt,
                             const float* __restrict__ bias, unsigned short* __restrict__ out) {
  const int H = 256, W = 256;
  long long tid = blockIdx.x * 256LL + threadIdx.x;     // P*16 threads
  int og = (int)(tid & 15);
  long long pix = tid >> 4;
  int HW = H * W;
  int b = (int)(pix / HW);
  int rem = (int)(pix - (long long)b * HW);
  int y = rem / W, x = rem - (rem / W) * W;
  int o0 = og * 4;
  float acc0 = bias[o0], acc1 = bias[o0 + 1], acc2 = bias[o0 + 2], acc3 = bias[o0 + 3];
  for (int tap = 0; tap < 9; ++tap) {
    int dy = tap / 3 - 1, dx = tap % 3 - 1;
    int ys = y + dy, xs = x + dx;
    if ((unsigned)ys >= (unsigned)H || (unsigned)xs >= (unsigned)W) continue;
    for (int c = 0; c < 3; ++c) {
      float xv = in[((long long)(b * 3 + c) * H + ys) * W + xs];
      float w0 = Wt[((long long)(o0 + 0) * 3 + c) * 9 + tap];
      float w1 = Wt[((long long)(o0 + 1) * 3 + c) * 9 + tap];
      float w2 = Wt[((long long)(o0 + 2) * 3 + c) * 9 + tap];
      float w3 = Wt[((long long)(o0 + 3) * 3 + c) * 9 + tap];
      acc0 = fmaf(w0, xv, acc0); acc1 = fmaf(w1, xv, acc1);
      acc2 = fmaf(w2, xv, acc2); acc3 = fmaf(w3, xv, acc3);
    }
  }
  unsigned short* op = out + pix * 64 + o0;
  op[0] = f2bf(fmaxf(acc0, 0.f)); op[1] = f2bf(fmaxf(acc1, 0.f));
  op[2] = f2bf(fmaxf(acc2, 0.f)); op[3] = f2bf(fmaxf(acc3, 0.f));
}

// ---------- WMMA implicit-GEMM conv3x3 SAME, bf16 NHWC in/out, relu+bias ----------
// Wave tile: 16 pixels x 32 couts (one A fragment feeds two WMMAs).
// Block: 8 waves = 4 pixel-groups x 2 cout32-groups => 64 pixels x 64 couts.
// Per-tap 64xCIN weight tile staged in LDS (async load-to-LDS when available).
template <int CIN, int COUT, int H>
__global__ void conv_wmma_k(const unsigned short* __restrict__ in,
                            const unsigned short* __restrict__ wb,
                            const float* __restrict__ bias,
                            unsigned short* __restrict__ out) {
  constexpr int W = H;
  constexpr int HW = H * W;
  constexpr int NCG = CIN >> 5;
  __shared__ unsigned short smw[64 * CIN];

  const int lane = threadIdx.x & 31;
  const int wv = threadIdx.x >> 5;                     // 0..7
  const int pixGroup = blockIdx.x * 4 + (wv & 3);
  const int coutSub = (wv >> 2) * 32;                  // 0 or 32 within block tile
  const int n = lane & 15;                             // A row / B col / D col
  const int khalf = lane >> 4;
  const int pix = pixGroup * 16 + n;
  const int bimg = pix / HW;
  const int rem = pix - bimg * HW;
  const int y = rem / W;
  const int x = rem - y * W;

  v8f acc0 = {};
  v8f acc1 = {};
  union Frag { unsigned u[8]; v16bf v; };
  int koffA[8], koffB[8];
#pragma unroll
  for (int v = 0; v < 8; ++v) {
    koffA[v] = ((v >> 2) << 4) + (khalf << 3) + ((v & 3) << 1); // ISA 16-bit A 16x32 layout
    koffB[v] = ((v >> 2) << 4) + (khalf << 3) + ((v & 3) << 1); // same k order as A (16-bit A 16x32 layout)
  }
  const unsigned short* wblk = wb + (long long)blockIdx.y * 64 * CIN;
  const unsigned short* ws0 = &smw[(coutSub + n) * CIN];
  const unsigned short* ws1 = ws0 + 16 * CIN;

  for (int tap = 0; tap < 9; ++tap) {
    const int dy = tap / 3 - 1, dx = tap % 3 - 1;
    const int ys = y + dy, xs = x + dx;
    const bool valid = (unsigned)ys < (unsigned)H && (unsigned)xs < (unsigned)W;
    const long long abase = valid ? (long long)((bimg * H + ys) * W + xs) * CIN : 0;

    // stage this tap's 64 x CIN bf16 weight tile into LDS
    const unsigned short* wt = wblk + (long long)tap * (COUT * CIN);
    constexpr int CHUNKS = (64 * CIN) / 8;             // 16-byte chunks; multiple of 256
    __syncthreads();                                   // previous tap's reads complete
#if __has_builtin(__builtin_amdgcn_global_load_async_to_lds_b128)
    for (int i = threadIdx.x; i < CHUNKS; i += 256) {
      __builtin_amdgcn_global_load_async_to_lds_b128(
          (gv4p)(wt + i * 8), (lv4p)(&smw[i * 8]), 0, 0);
    }
#if __has_builtin(__builtin_amdgcn_s_wait_asynccnt)
    __builtin_amdgcn_s_wait_asynccnt(0);
#else
    asm volatile("s_wait_asynccnt 0" ::: "memory");
#endif
#else
    for (int i = threadIdx.x; i < CHUNKS; i += 256) {
      *(uint4*)(&smw[i * 8]) = *(const uint4*)(wt + i * 8);
    }
#endif
    __syncthreads();                                   // tile visible to all waves

#pragma unroll 4
    for (int cg = 0; cg < NCG; ++cg) {
      const int cbase = cg << 5;
      Frag A, B0, B1;
#pragma unroll
      for (int v = 0; v < 8; ++v) {
        A.u[v]  = valid ? *(const unsigned*)(in + abase + cbase + koffA[v]) : 0u;
        B0.u[v] = *(const unsigned*)(ws0 + cbase + koffB[v]);
        B1.u[v] = *(const unsigned*)(ws1 + cbase + koffB[v]);
      }
      acc0 = __builtin_amdgcn_wmma_f32_16x16x32_bf16(false, A.v, false, B0.v,
                                                     (short)0, acc0, false, false);
      acc1 = __builtin_amdgcn_wmma_f32_16x16x32_bf16(false, A.v, false, B1.v,
                                                     (short)0, acc1, false, false);
    }
  }
  const int coutBase = blockIdx.y * 64 + coutSub;
  const int co0 = coutBase + n;
  const int co1 = co0 + 16;
  const float bv0 = bias[co0];
  const float bv1 = bias[co1];
#pragma unroll
  for (int v = 0; v < 8; ++v) {
    int opix = pixGroup * 16 + khalf * 8 + v;          // ISA f32 C/D 16x16 layout
    float r0 = acc0[v] + bv0;
    float r1 = acc1[v] + bv1;
    out[(long long)opix * COUT + co0] = f2bf(r0 > 0.f ? r0 : 0.f);
    out[(long long)opix * COUT + co1] = f2bf(r1 > 0.f ? r1 : 0.f);
  }
}

// ---------- maxpool 2x2 on bf16 NHWC ----------
__global__ void maxpool_k(const unsigned short* __restrict__ in, unsigned short* __restrict__ out,
                          int H, int W, int C, long long total) {
  long long tid = blockIdx.x * 256LL + threadIdx.x;
  if (tid >= total) return;
  int OW = W / 2, OH = H / 2;
  int c = (int)(tid % C);
  long long r = tid / C;
  int x = (int)(r % OW); r /= OW;
  int y = (int)(r % OH);
  int b = (int)(r / OH);
  long long base = ((long long)(b * H + 2 * y) * W + 2 * x) * C + c;
  float m = bf2f(in[base]);
  m = fmaxf(m, bf2f(in[base + C]));
  m = fmaxf(m, bf2f(in[base + (long long)W * C]));
  m = fmaxf(m, bf2f(in[base + (long long)W * C + C]));
  out[((long long)(b * OH + y) * OW + x) * C + c] = f2bf(m);
}

// ---------- perceptual MSE over final features (bf16) ----------
__global__ void perc_k(const unsigned short* __restrict__ a, const unsigned short* __restrict__ b,
                       long long n, float* S) {
  long long tid = blockIdx.x * 256LL + threadIdx.x;
  float v = 0.f;
  if (tid < n) { float d = bf2f(a[tid]) - bf2f(b[tid]); v = d * d; }
  float r = block_reduce256(v);
  if (threadIdx.x == 0) atomicAdd(&S[14], r);
}

// ---------- smooth L1 + MSE ----------
__global__ void elem_k(const float* __restrict__ t, const float* __restrict__ p,
                       long long n, float* S) {
  long long tid = blockIdx.x * 256LL + threadIdx.x;
  float sm = 0.f, ms = 0.f;
  if (tid < n) {
    float d = t[tid] - p[tid];
    float ad = fabsf(d);
    sm = ad < 1.f ? 0.5f * d * d : ad - 0.5f;
    ms = d * d;
  }
  float r0 = block_reduce256(sm);
  float r1 = block_reduce256(ms);
  if (threadIdx.x == 0) { atomicAdd(&S[0], r0); atomicAdd(&S[1], r1); }
}

// ---------- OKLab + CIEDE2000 ----------
static __device__ inline float cbrt_eps(float t) {
  float s = t < 0.f ? -1.f : 1.f;
  return s * powf(fabsf(t) + 1e-6f, 1.0f / 3.0f);
}
static __device__ inline void oklab(float r, float g, float b, float* L, float* a, float* bo) {
  float l = 0.4122214708f * r + 0.5363325363f * g + 0.0514459929f * b;
  float m = 0.2119034982f * r + 0.6806995451f * g + 0.1073969566f * b;
  float s = 0.0883024619f * r + 0.2817188376f * g + 0.6299787005f * b;
  float l_ = cbrt_eps(l), m_ = cbrt_eps(m), s_ = cbrt_eps(s);
  *L  = 0.2104542553f * l_ + 0.793617785f * m_ - 0.0040720468f * s_;
  *a  = 1.9779984951f * l_ - 2.428592205f * m_ + 0.4505937099f * s_;
  *bo = 0.0259040371f * l_ + 0.7827717662f * m_ - 0.808675766f * s_;
}
static __device__ inline float cosd(float x) { return cosf(x * 0.017453292519943295f); }
static __device__ inline float ciede2000_dev(float L1, float a1, float b1,
                                             float L2, float a2, float b2) {
  const float D2R = 0.017453292519943295f;
  const float p25 = 6103515625.0f; // 25^7
  float C1 = hypotf(a1, b1), C2 = hypotf(a2, b2);
  float Cb = 0.5f * (C1 + C2);
  float c7 = powf(Cb, 7.0f);
  float G = 0.5f * (1.0f - sqrtf(c7 / (c7 + p25)));
  float a1p = (1.f + G) * a1, a2p = (1.f + G) * a2;
  float C1p = hypotf(a1p, b1), C2p = hypotf(a2p, b2);
  float h1p = fmodf(atan2f(b1, a1p) * 57.29577951308232f, 360.0f); if (h1p < 0.f) h1p += 360.f;
  float h2p = fmodf(atan2f(b2, a2p) * 57.29577951308232f, 360.0f); if (h2p < 0.f) h2p += 360.f;
  float dLp = L2 - L1, dCp = C2p - C1p;
  float hd = h2p - h1p;
  float dhp = fabsf(hd) <= 180.f ? hd : (hd > 180.f ? hd - 360.f : hd + 360.f);
  bool zero = (C1p * C2p == 0.f);
  if (zero) dhp = 0.f;
  float dHp = 2.0f * sqrtf(C1p * C2p) * sinf(D2R * dhp * 0.5f);
  float Lb = 0.5f * (L1 + L2), Cbp = 0.5f * (C1p + C2p);
  float hsum = h1p + h2p;
  float hbar = fabsf(h1p - h2p) <= 180.f ? 0.5f * hsum
             : (hsum < 360.f ? 0.5f * (hsum + 360.f) : 0.5f * (hsum - 360.f));
  if (zero) hbar = hsum;
  float T = 1.0f - 0.17f * cosd(hbar - 30.f) + 0.24f * cosd(2.f * hbar)
          + 0.32f * cosd(3.f * hbar + 6.f) - 0.2f * cosd(4.f * hbar - 63.f);
  float hq = (hbar - 275.f) * 0.04f;
  float dth = 30.f * expf(-hq * hq);
  float cb7 = powf(Cbp, 7.0f);
  float Rc = 2.0f * sqrtf(cb7 / (cb7 + p25));
  float L50 = (Lb - 50.f) * (Lb - 50.f);
  float Sl = 1.0f + 0.015f * L50 / sqrtf(20.0f + L50);
  float Sc = 1.0f + 0.045f * Cbp;
  float Sh = 1.0f + 0.015f * Cbp * T;
  float Rt = -sinf(D2R * 2.0f * dth) * Rc;
  float q1 = dLp / Sl, q2 = dCp / Sc, q3 = dHp / Sh;
  return sqrtf(q1 * q1 + q2 * q2 + q3 * q3 + Rt * q2 * q3);
}
__global__ void color_k(const float* __restrict__ yt, const float* __restrict__ yp, float* S) {
  const int HW = 65536;
  long long tid = blockIdx.x * 256LL + threadIdx.x;   // 2*HW pixels
  float cl = 0.f, de = 0.f;
  if (tid < 2LL * HW) {
    int b = (int)(tid / HW), p = (int)(tid % HW);
    const float* t = yt + (long long)b * 3 * HW + p;
    const float* q = yp + (long long)b * 3 * HW + p;
    float L1, A1, B1, L2, A2, B2;
    oklab(t[0], t[HW], t[2 * HW], &L1, &A1, &B1);
    oklab(q[0], q[HW], q[2 * HW], &L2, &A2, &B2);
    cl = fabsf(L1 - L2) + fabsf(A1 - A2) + fabsf(B1 - B2);
    de = ciede2000_dev(L1, A1, B1, L2, A2, B2);
  }
  float r0 = block_reduce256(cl);
  float r1 = block_reduce256(de);
  if (threadIdx.x == 0) { atomicAdd(&S[2], r0); atomicAdd(&S[3], r1); }
}

// ---------- soft histogram: bin-per-thread, data staged through LDS ----------
__global__ void hist_k(const float* __restrict__ x, long long n, float* hist) {
  __shared__ float cache[256];
  int tid = threadIdx.x;
  float e = tid * (1.0f / 255.0f);
  float acc = 0.f;
  for (long long base = blockIdx.x * 256LL; base < n; base += gridDim.x * 256LL) {
    long long i = base + tid;
    cache[tid] = (i < n) ? x[i] : 1.0e9f;  // sentinel -> exp underflows to 0
    __syncthreads();
    for (int k = 0; k < 256; ++k) {
      float d = (cache[k] - e) * 100.0f;
      acc += expf(-0.5f * d * d);
    }
    __syncthreads();
  }
  atomicAdd(&hist[tid], acc);
}

// ---------- fused SSIM (per-scale, per-(b,c) sums) ----------
__global__ void ssim_k(const float* __restrict__ X, const float* __restrict__ Y,
                       int H, int W, int scale, float* S) {
  int bc = blockIdx.y;                   // 0..5  (B*C)
  int OH = H - 10, OW = W - 10;
  long long tid = blockIdx.x * 256LL + threadIdx.x;
  float sv = 0.f, cv = 0.f;
  if (tid < (long long)OH * OW) {
    int ox = (int)(tid % OW), oy = (int)(tid / OW);
    float g[11]; float gs = 0.f;
#pragma unroll
    for (int i = 0; i < 11; ++i) { float c = (float)i - 5.0f; g[i] = expf(-c * c / 4.5f); gs += g[i]; }
#pragma unroll
    for (int i = 0; i < 11; ++i) g[i] /= gs;
    const float* Xp = X + (long long)bc * H * W;
    const float* Yp = Y + (long long)bc * H * W;
    float mu1 = 0, mu2 = 0, xx = 0, yy = 0, xy = 0;
    for (int i = 0; i < 11; ++i) {
      float r1 = 0, r2 = 0, rxx = 0, ryy = 0, rxy = 0;
      const float* Xr = Xp + (long long)(oy + i) * W + ox;
      const float* Yr = Yp + (long long)(oy + i) * W + ox;
#pragma unroll
      for (int j = 0; j < 11; ++j) {
        float wv = g[j];
        float a = Xr[j], b = Yr[j];
        r1 += wv * a; r2 += wv * b;
        rxx += wv * a * a; ryy += wv * b * b; rxy += wv * a * b;
      }
      mu1 += g[i] * r1; mu2 += g[i] * r2;
      xx += g[i] * rxx; yy += g[i] * ryy; xy += g[i] * rxy;
    }
    const float C1 = 1e-4f, C2 = 9e-4f;
    float mu1s = mu1 * mu1, mu2s = mu2 * mu2, mu12 = mu1 * mu2;
    float s1 = xx - mu1s, s2 = yy - mu2s, s12 = xy - mu12;
    cv = (2.f * s12 + C2) / (s1 + s2 + C2);
    sv = ((2.f * mu12 + C1) / (mu1s + mu2s + C1)) * cv;
  }
  float cr = block_reduce256(cv);
  float sr = block_reduce256(sv);
  if (threadIdx.x == 0) {
    atomicAdd(&S[16 + scale * 12 + bc], cr);
    atomicAdd(&S[16 + scale * 12 + 6 + bc], sr);
  }
}

// ---------- avgpool 2x2 on f32 "planes" (B*C=6) ----------
__global__ void avgpool_k(const float* __restrict__ in, float* __restrict__ out, int H, int W) {
  int OH = H / 2, OW = W / 2;
  long long total = 6LL * OH * OW;
  long long tid = blockIdx.x * 256LL + threadIdx.x;
  if (tid >= total) return;
  int x = (int)(tid % OW);
  long long r = tid / OW;
  int y = (int)(r % OH);
  int p = (int)(r / OH);
  const float* ip = in + (long long)p * H * W;
  float s = ip[(2 * y) * (long long)W + 2 * x] + ip[(2 * y) * (long long)W + 2 * x + 1]
          + ip[(2 * y + 1) * (long long)W + 2 * x] + ip[(2 * y + 1) * (long long)W + 2 * x + 1];
  out[(long long)p * OH * OW + (long long)y * OW + x] = 0.25f * s;
}

// ---------- final combine ----------
__global__ void combine_k(float* S, float* out) {
  int t = threadIdx.x;
  float ht = S[128 + t], hp = S[384 + t];
  float sht = block_reduce256(ht);
  float shp = block_reduce256(hp);
  float dt = fabsf(ht / (sht + 1e-8f) - hp / (shp + 1e-8f));
  float hist_l = block_reduce256(dt) * (1.0f / 256.0f);
  if (t == 0) {
    const float N = 2.0f * 3.0f * 256.0f * 256.0f;
    float sm = S[0] / N;
    float mse = S[1] / N;
    float psnr_l = 40.0f + 10.0f * log10f(mse + 1e-8f);
    float color_l = S[2] / N;
    float de = S[3] / (2.0f * 256.0f * 256.0f);
    float perc = S[14] / 262144.0f;
    const float wms[5] = {0.0448f, 0.2856f, 0.3001f, 0.2363f, 0.1333f};
    const float cnt[5] = {60516.f, 13924.f, 2916.f, 484.f, 36.f};
    float msum = 0.f;
    for (int j = 0; j < 6; ++j) {
      float prod = 1.f;
      for (int i = 0; i < 5; ++i) {
        float v = (i < 4 ? S[16 + i * 12 + j] : S[16 + i * 12 + 6 + j]) / cnt[i];
        v = v > 0.f ? v : 0.f;
        prod *= powf(v, wms[i]);
      }
      msum += prod;
    }
    float ms_l = 1.0f - msum / 6.0f;
    out[0] = 0.8f * sm + 0.3f * perc + 0.1f * hist_l + 0.002f * psnr_l
           + 0.5f * color_l + 0.8f * ms_l + 0.2f * de;
  }
}

// ============================ host orchestration ============================
extern "C" void kernel_launch(void* const* d_in, const int* in_sizes, int n_in,
                              void* d_out, int out_size, void* d_ws, size_t ws_size,
                              hipStream_t stream) {
  (void)in_sizes; (void)n_in; (void)out_size; (void)ws_size;
  const float* y_true = (const float*)d_in[0];
  const float* y_pred = (const float*)d_in[1];
  char* ws = (char*)d_ws;
  float* S               = (float*)ws;                              // 1024 floats
  unsigned short* featT  = (unsigned short*)(ws + (16ull << 10));   // 512 KB
  unsigned short* wpack  = (unsigned short*)(ws + (1ull << 20));    // ~40 MB
  unsigned short* actA   = (unsigned short*)(ws + (48ull << 20));   // 17 MB
  unsigned short* actB   = (unsigned short*)(ws + (68ull << 20));   // 17 MB
  float* sx0 = (float*)(ws + (88ull << 20));
  float* sy0 = (float*)(ws + (90ull << 20));
  float* sx1 = (float*)(ws + (92ull << 20));
  float* sy1 = (float*)(ws + (94ull << 20));

  zero_k<<<1, 256, 0, stream>>>(S);

  static const int CIN[16]  = {3, 64, 64, 128, 128, 256, 256, 256, 256, 512, 512, 512, 512, 512, 512, 512};
  static const int COUT[16] = {64, 64, 128, 128, 256, 256, 256, 256, 512, 512, 512, 512, 512, 512, 512, 512};
  static const int HIN[16]  = {256, 256, 128, 128, 64, 64, 64, 64, 32, 32, 32, 32, 16, 16, 16, 16};

  long long woff[16]; long long wacc = 0;
  for (int l = 0; l < 16; ++l) { woff[l] = wacc; wacc += 9LL * CIN[l] * COUT[l]; }
  for (int l = 1; l < 16; ++l) {
    long long tot = 9LL * CIN[l] * COUT[l];
    pack_w_k<<<(int)((tot + 255) / 256), 256, 0, stream>>>(
        (const float*)d_in[2 + 2 * l], wpack + woff[l], CIN[l], COUT[l], tot);
  }

#define LAUNCH_CONV(CINV, COUTV, HV)                                           \
  conv_wmma_k<CINV, COUTV, HV>                                                 \
      <<<dim3((2 * HV * HV) / 64, COUTV / 64), 256, 0, stream>>>(              \
          cur, wpack + woff[l], (const float*)d_in[3 + 2 * l], oth)

  for (int img = 0; img < 2; ++img) {
    const float* x = (img == 0) ? y_true : y_pred;
    conv_first_k<<<8192, 256, 0, stream>>>(x, (const float*)d_in[2], (const float*)d_in[3], actA);
    unsigned short* cur = actA;
    unsigned short* oth = actB;
    for (int l = 1; l < 16; ++l) {
      if      (l == 1)  LAUNCH_CONV(64, 64, 256);
      else if (l == 2)  LAUNCH_CONV(64, 128, 128);
      else if (l == 3)  LAUNCH_CONV(128, 128, 128);
      else if (l == 4)  LAUNCH_CONV(128, 256, 64);
      else if (l <= 7)  LAUNCH_CONV(256, 256, 64);
      else if (l == 8)  LAUNCH_CONV(256, 512, 32);
      else if (l <= 11) LAUNCH_CONV(512, 512, 32);
      else              LAUNCH_CONV(512, 512, 16);
      { unsigned short* tp = cur; cur = oth; oth = tp; }
      if (l == 1 || l == 3 || l == 7 || l == 11) {
        int H = HIN[l];
        int OH = H / 2;
        long long tot = 2LL * OH * OH * COUT[l];
        maxpool_k<<<(int)((tot + 255) / 256), 256, 0, stream>>>(cur, oth, H, H, COUT[l], tot);
        { unsigned short* tp = cur; cur = oth; oth = tp; }
      }
    }
    if (img == 0) {
      (void)hipMemcpyAsync(featT, cur, 262144ull * 2, hipMemcpyDeviceToDevice, stream);
    } else {
      perc_k<<<1024, 256, 0, stream>>>(featT, cur, 262144LL, S);
    }
  }
#undef LAUNCH_CONV

  elem_k<<<1536, 256, 0, stream>>>(y_true, y_pred, 393216LL, S);
  color_k<<<512, 256, 0, stream>>>(y_true, y_pred, S);
  hist_k<<<256, 256, 0, stream>>>(y_true, 393216LL, S + 128);
  hist_k<<<256, 256, 0, stream>>>(y_pred, 393216LL, S + 384);

  ssim_k<<<dim3((246 * 246 + 255) / 256, 6), 256, 0, stream>>>(y_true, y_pred, 256, 256, 0, S);
  avgpool_k<<<(6 * 128 * 128 + 255) / 256, 256, 0, stream>>>(y_true, sx0, 256, 256);
  avgpool_k<<<(6 * 128 * 128 + 255) / 256, 256, 0, stream>>>(y_pred, sy0, 256, 256);
  ssim_k<<<dim3((118 * 118 + 255) / 256, 6), 256, 0, stream>>>(sx0, sy0, 128, 128, 1, S);
  avgpool_k<<<(6 * 64 * 64 + 255) / 256, 256, 0, stream>>>(sx0, sx1, 128, 128);
  avgpool_k<<<(6 * 64 * 64 + 255) / 256, 256, 0, stream>>>(sy0, sy1, 128, 128);
  ssim_k<<<dim3((54 * 54 + 255) / 256, 6), 256, 0, stream>>>(sx1, sy1, 64, 64, 2, S);
  avgpool_k<<<(6 * 32 * 32 + 255) / 256, 256, 0, stream>>>(sx1, sx0, 64, 64);
  avgpool_k<<<(6 * 32 * 32 + 255) / 256, 256, 0, stream>>>(sy1, sy0, 64, 64);
  ssim_k<<<dim3((22 * 22 + 255) / 256, 6), 256, 0, stream>>>(sx0, sy0, 32, 32, 3, S);
  avgpool_k<<<(6 * 16 * 16 + 255) / 256, 256, 0, stream>>>(sx0, sx1, 32, 32);
  avgpool_k<<<(6 * 16 * 16 + 255) / 256, 256, 0, stream>>>(sy0, sy1, 32, 32);
  ssim_k<<<dim3(1, 6), 256, 0, stream>>>(sx1, sy1, 16, 16, 4, S);

  combine_k<<<1, 256, 0, stream>>>(S, (float*)d_out);
}
